// MultiHeadAttention_62483184222889
// MI455X (gfx1250) — matrix operand order repaired and checked by reading, not compiled, not run on hardware
//
#include <hip/hip_runtime.h>
#include <stdint.h>

typedef __attribute__((ext_vector_type(16))) __bf16 v16bf;
typedef __attribute__((ext_vector_type(8)))  float  v8f;

#define B_  2
#define T_  2048
#define E_  1024
#define H_  16
#define D_  64
#define BT_ (B_ * T_)
#define N3E (3 * E_)

__device__ __forceinline__ unsigned short f32_bf16(float f) {
  union { float f; unsigned u; } v; v.f = f;
  unsigned r = v.u + 0x7FFFu + ((v.u >> 16) & 1u);
  return (unsigned short)(r >> 16);
}

union FragU { v16bf v; uint4 q[2]; unsigned short s[16]; };

__device__ __forceinline__ v8f wmma_bf16(v16bf a, v16bf b, v8f c) {
  // D = A(16x32 bf16) * B(32x16 bf16) + C(16x16 f32)
  return __builtin_amdgcn_wmma_f32_16x16x32_bf16(
      /*neg_a=*/false, a, /*neg_b=*/false, b,
      /*c_mod=*/(short)0, c, /*reuse_a=*/false, /*reuse_b=*/false);
}

// CDNA5 async global->LDS copy (16B). offset: immediate applies to BOTH the
// LDS destination address and the global address (ISA 15.18.3 pseudocode).
__device__ __forceinline__ void async_ldscpy_b128(unsigned lds_addr, unsigned gvoff,
                                                  const void* sbase) {
  asm volatile("global_load_async_to_lds_b128 %0, %1, %2"
               :: "v"(lds_addr), "v"(gvoff), "s"(sbase) : "memory");
}
__device__ __forceinline__ void async_ldscpy_b128_off16(unsigned lds_addr, unsigned gvoff,
                                                        const void* sbase) {
  asm volatile("global_load_async_to_lds_b128 %0, %1, %2 offset:16"
               :: "v"(lds_addr), "v"(gvoff), "s"(sbase) : "memory");
}
// Async ops complete in order: waiting <=4 retires the older tile's 4 copies
// while the just-issued tile's 4 stay in flight behind the WMMAs.
__device__ __forceinline__ void wait_async0() {
  asm volatile("s_wait_asynccnt 0" ::: "memory");
}
__device__ __forceinline__ void wait_async4() {
  asm volatile("s_wait_asynccnt 4" ::: "memory");
}

// ---------------------------------------------------------------- convert
// vectorized f32 -> bf16 (n % 4 == 0 for all call sites)
__global__ void k_f32_to_bf16(const float* __restrict__ in,
                              unsigned short* __restrict__ out, int n) {
  int i = (blockIdx.x * blockDim.x + threadIdx.x) * 4;
  if (i < n) {
    float4 f = *(const float4*)(in + i);
    unsigned lo = (unsigned)f32_bf16(f.x) | ((unsigned)f32_bf16(f.y) << 16);
    unsigned hi = (unsigned)f32_bf16(f.z) | ((unsigned)f32_bf16(f.w) << 16);
    *(uint2*)(out + i) = make_uint2(lo, hi);
  }
}

// transpose + convert: in f32 [K][N] -> out bf16 [N][K]  (K%32==0, N%32==0)
__global__ __launch_bounds__(256) void k_transpose_bf16(
    const float* __restrict__ in, unsigned short* __restrict__ out, int K, int N) {
  __shared__ float tile[32][33];
  const int k0 = blockIdx.y * 32, n0 = blockIdx.x * 32;
  const int tx = threadIdx.x & 31, ty = threadIdx.x >> 5;   // ty 0..7
#pragma unroll
  for (int i = 0; i < 4; ++i) {
    int k = ty + i * 8;
    tile[k][tx] = in[(size_t)(k0 + k) * N + n0 + tx];
  }
  __syncthreads();
#pragma unroll
  for (int i = 0; i < 4; ++i) {
    int n = ty + i * 8;
    out[(size_t)(n0 + n) * K + k0 + tx] = f32_bf16(tile[tx][n]);
  }
}

// ---------------------------------------------------------------- GEMM
// C[M,N] = A_bf16[M,K] @ Bt_bf16[N,K]^T + bias[N]   (M%128, N%128, K%32 == 0)
// 256 threads = 8 waves; block tile 128x128; K-tile 32.
// Double-buffered async global->LDS staging pipelined on ASYNCcnt.
__global__ __launch_bounds__(256) void k_gemm_bf16(
    const unsigned short* __restrict__ A,    // [M][K]
    const unsigned short* __restrict__ Bt,   // [N][K] (pre-transposed weights)
    const float* __restrict__ bias, float* __restrict__ C,
    int M, int N, int K) {
  __shared__ unsigned short As[2][128 * 32];   // [buf][m][k]
  __shared__ unsigned short Bs[2][128 * 32];   // [buf][n][k]

  const int tid  = threadIdx.x;
  const int lane = tid & 31;
  const int wid  = tid >> 5;
  const int wm   = wid >> 2;          // 0..1
  const int wn   = wid & 3;           // 0..3
  const int m0   = blockIdx.y * 128;
  const int n0   = blockIdx.x * 128;
  const int ln   = lane & 15;
  const int klo  = (lane >> 4) * 8;   // A-fragment chunk base (elements)
  const int kbb  = (lane >> 4) * 16;  // unused: B fragments also read at klo

  const int row = tid >> 1;           // 0..127 (tile row for staging)
  const int seg = (tid & 1) * 16;     // element offset 0 / 16

  const unsigned ldsA0 = (unsigned)(uintptr_t)&As[0][row * 32 + seg];
  const unsigned ldsA1 = (unsigned)(uintptr_t)&As[1][row * 32 + seg];
  const unsigned ldsB0 = (unsigned)(uintptr_t)&Bs[0][row * 32 + seg];
  const unsigned ldsB1 = (unsigned)(uintptr_t)&Bs[1][row * 32 + seg];
  const unsigned gaBase = (unsigned)(((m0 + row) * K + seg) * 2);   // byte offset
  const unsigned gbBase = (unsigned)(((n0 + row) * K + seg) * 2);

  const int ntiles = K / 32;

  // prologue: stage tile 0 into buffer 0
  async_ldscpy_b128(ldsA0, gaBase, A);
  async_ldscpy_b128_off16(ldsA0, gaBase, A);
  async_ldscpy_b128(ldsB0, gbBase, Bt);
  async_ldscpy_b128_off16(ldsB0, gbBase, Bt);

  v8f acc[4][2] = {};

  for (int i = 0; i < ntiles; ++i) {
    const int cur = i & 1;
    if (i + 1 < ntiles) {
      // stage tile i+1 into the other buffer (consumed & barriered in iter i-1)
      const unsigned nga = gaBase + (unsigned)((i + 1) * 64);
      const unsigned ngb = gbBase + (unsigned)((i + 1) * 64);
      const unsigned la = cur ? ldsA0 : ldsA1;
      const unsigned lb = cur ? ldsB0 : ldsB1;
      async_ldscpy_b128(la, nga, A);
      async_ldscpy_b128_off16(la, nga, A);
      async_ldscpy_b128(lb, ngb, Bt);
      async_ldscpy_b128_off16(lb, ngb, Bt);
      wait_async4();   // tile i complete; tile i+1 still in flight
    } else {
      wait_async0();
    }
    __syncthreads();

    const unsigned short* Asb = As[cur];
    const unsigned short* Bsb = Bs[cur];
    v16bf afr[4], bfr[2];
#pragma unroll
    for (int sm = 0; sm < 4; ++sm) {
      int r = wm * 64 + sm * 16 + ln;
      FragU f;
      f.q[0] = *(const uint4*)(&Asb[r * 32 + klo]);
      f.q[1] = *(const uint4*)(&Asb[r * 32 + klo + 16]);
      afr[sm] = f.v;
    }
#pragma unroll
    for (int sn = 0; sn < 2; ++sn) {
      int c = wn * 32 + sn * 16 + ln;
      FragU f;
      f.q[0] = *(const uint4*)(&Bsb[c * 32 + klo]);
      f.q[1] = *(const uint4*)(&Bsb[c * 32 + klo + 16]);
      bfr[sn] = f.v;
    }
#pragma unroll
    for (int sm = 0; sm < 4; ++sm)
#pragma unroll
      for (int sn = 0; sn < 2; ++sn)
        acc[sm][sn] = wmma_bf16(afr[sm], bfr[sn], acc[sm][sn]);
    __syncthreads();
  }

  const int rbase = (lane >> 4) * 8;   // C-layout: lanes16-31 hold rows 8..15
#pragma unroll
  for (int sm = 0; sm < 4; ++sm) {
#pragma unroll
    for (int sn = 0; sn < 2; ++sn) {
      int col = n0 + wn * 32 + sn * 16 + ln;
      float bv = bias[col];
#pragma unroll
      for (int r = 0; r < 8; ++r) {
        int rw = m0 + wm * 64 + sm * 16 + r + rbase;
        C[(size_t)rw * N + col] = acc[sm][sn][r] + bv;
      }
    }
  }
}

// ---------------------------------------------------------------- RoPE + pack
// qkv f32 [BT][3E] -> q16,k16 [B,H,T,D] bf16 (rotary applied), vT16 [B,H,D,T] bf16
__global__ void k_rope_pack(const float* __restrict__ qkv,
                            unsigned short* __restrict__ q16,
                            unsigned short* __restrict__ k16,
                            unsigned short* __restrict__ vT16) {
  int tid = blockIdx.x * blockDim.x + threadIdx.x;
  if (tid >= B_ * T_ * E_) return;
  int j  = tid & 63;
  int h  = (tid >> 6) & 15;
  int bt = tid >> 10;
  int b  = bt >> 11;
  int t  = bt & 2047;

  const float* row = qkv + (size_t)bt * N3E;
  int base = h * 64;
  float qv = row[base + j];
  float kv = row[E_ + base + j];
  float vv = row[2 * E_ + base + j];
  float qp = row[base + (j ^ 32)];
  float kp = row[E_ + base + (j ^ 32)];
  float sgn = (j < 32) ? -1.f : 1.f;
  // inv_freq = 10000^(-(j%32)/32);  log(10000)/32 = 0.2878231366242557
  float invf = __expf(-(float)(j & 31) * 0.2878231366242557f);
  float ang = (float)t * invf;
  float c = cosf(ang), s = sinf(ang);
  float qo = qv * c + sgn * qp * s;
  float ko = kv * c + sgn * kp * s;

  size_t bh = (size_t)(b * H_ + h);
  q16[(bh * T_ + t) * D_ + j]  = f32_bf16(qo);
  k16[(bh * T_ + t) * D_ + j]  = f32_bf16(ko);
  vT16[(bh * D_ + j) * T_ + t] = f32_bf16(vv);
}

// ---------------------------------------------------------------- flash attention
// 4 waves / block; each wave owns one 16-query tile of the same (b,h).
// All waves run the block-uniform key-loop bound so barriers stay uniform;
// fully-masked tail tiles contribute 0 (exp(-inf)) and alpha=1.
__global__ __launch_bounds__(128) void k_flash_attn(
    const unsigned short* __restrict__ q16,
    const unsigned short* __restrict__ k16,
    const unsigned short* __restrict__ vT16,
    float* __restrict__ outp) {
  __shared__ unsigned short Pl[4][16 * 32];   // per-wave P transpose staging

  const int lane = threadIdx.x & 31;
  const int wid  = threadIdx.x >> 5;          // 0..3
  const int qb   = blockIdx.x * 64;
  const int q0   = qb + wid * 16;
  const int h    = blockIdx.y;
  const int b    = blockIdx.z;
  const size_t bh = (size_t)(b * H_ + h);

  const int ln  = lane & 15;
  const int klo = (lane >> 4) * 8;    // A-fragment chunk base
  const int kbb = (lane >> 4) * 16;   // unused: K and V fragments also read at klo
  const int mof = (lane >> 4) * 8;    // C-layout row offset for this half-wave

  // Q A-fragments: rows q0+ln, contraction d=0..31 (qa0) and d=32..63 (qa1)
  const unsigned short* qrow = q16 + (bh * T_ + q0 + ln) * D_;
  FragU qa0, qa1;
  qa0.q[0] = *(const uint4*)(qrow + klo);
  qa0.q[1] = *(const uint4*)(qrow + klo + 16);
  qa1.q[0] = *(const uint4*)(qrow + 32 + klo);
  qa1.q[1] = *(const uint4*)(qrow + 32 + klo + 16);

  v8f o[4] = {};
  v8f mrow, lrow;
#pragma unroll
  for (int r = 0; r < 8; ++r) { mrow[r] = -3.0e38f; lrow[r] = 0.f; }

  const int nblk = (qb + 64) / 32;    // uniform across the 4 waves
  const float scale = 0.125f;         // 1/sqrt(64)

  for (int blk = 0; blk < nblk; ++blk) {
    const int kb32 = blk * 32;
    v8f S[2];
#pragma unroll
    for (int t2 = 0; t2 < 2; ++t2) {
      // K^T B-fragments: key column = kb32 + t2*16 + ln, contraction over d
      const unsigned short* krow = k16 + (bh * T_ + kb32 + t2 * 16 + ln) * D_;
      FragU kf0, kf1;
      kf0.q[0] = *(const uint4*)(krow + klo);
      kf0.q[1] = *(const uint4*)(krow + klo + 16);
      kf1.q[0] = *(const uint4*)(krow + 32 + klo);
      kf1.q[1] = *(const uint4*)(krow + 32 + klo + 16);
      v8f sacc = {};
      sacc = wmma_bf16(qa0.v, kf0.v, sacc);
      sacc = wmma_bf16(qa1.v, kf1.v, sacc);
      S[t2] = sacc;
    }
    // scale + causal mask (C layout: vgpr r -> row r+mof, lane&15 -> col)
#pragma unroll
    for (int t2 = 0; t2 < 2; ++t2)
#pragma unroll
      for (int r = 0; r < 8; ++r) {
        int key = kb32 + t2 * 16 + ln;
        float sv = S[t2][r] * scale;
        S[t2][r] = (key <= q0 + r + mof) ? sv : -3.0e38f;
      }
    // online softmax; rows live in 16-lane halves under wave32
    v8f alpha;
#pragma unroll
    for (int r = 0; r < 8; ++r) {
      float v = fmaxf(S[0][r], S[1][r]);
      v = fmaxf(v, __shfl_xor(v, 1, 32));
      v = fmaxf(v, __shfl_xor(v, 2, 32));
      v = fmaxf(v, __shfl_xor(v, 4, 32));
      v = fmaxf(v, __shfl_xor(v, 8, 32));
      float mn = fmaxf(mrow[r], v);
      alpha[r] = __expf(mrow[r] - mn);
      mrow[r] = mn;
      S[0][r] = __expf(S[0][r] - mn);
      S[1][r] = __expf(S[1][r] - mn);
      float sum = S[0][r] + S[1][r];
      sum += __shfl_xor(sum, 1, 32);
      sum += __shfl_xor(sum, 2, 32);
      sum += __shfl_xor(sum, 4, 32);
      sum += __shfl_xor(sum, 8, 32);
      lrow[r] = lrow[r] * alpha[r] + sum;
    }
#pragma unroll
    for (int dt = 0; dt < 4; ++dt)
#pragma unroll
      for (int r = 0; r < 8; ++r) o[dt][r] *= alpha[r];

    // transpose P through this wave's LDS slab: C-layout -> A-fragment layout
    __syncthreads();
#pragma unroll
    for (int t2 = 0; t2 < 2; ++t2)
#pragma unroll
      for (int r = 0; r < 8; ++r)
        Pl[wid][(r + mof) * 32 + t2 * 16 + ln] = f32_bf16(S[t2][r]);
    __syncthreads();
    FragU pf;
    pf.q[0] = *(const uint4*)(&Pl[wid][ln * 32 + klo]);
    pf.q[1] = *(const uint4*)(&Pl[wid][ln * 32 + klo + 16]);

    // O += P(16x32) @ V(32x16) per 16-wide d-tile; V read from [B,H,D,T]
#pragma unroll
    for (int dt = 0; dt < 4; ++dt) {
      const unsigned short* vrow = vT16 + (bh * D_ + dt * 16 + ln) * T_ + kb32;
      FragU vf;
      vf.q[0] = *(const uint4*)(vrow + klo);
      vf.q[1] = *(const uint4*)(vrow + klo + 16);
      o[dt] = wmma_bf16(pf.v, vf.v, o[dt]);
    }
    __syncthreads();
  }

  // epilogue: divide by softmax denom, write merged-head layout [B*T][E]
#pragma unroll
  for (int dt = 0; dt < 4; ++dt)
#pragma unroll
    for (int r = 0; r < 8; ++r) {
      int rw  = b * T_ + q0 + r + mof;
      int col = h * 64 + dt * 16 + ln;
      outp[(size_t)rw * E_ + col] = o[dt][r] / lrow[r];
    }
}

// ---------------------------------------------------------------- launch
extern "C" void kernel_launch(void* const* d_in, const int* in_sizes, int n_in,
                              void* d_out, int out_size, void* d_ws, size_t ws_size,
                              hipStream_t stream) {
  const float* x     = (const float*)d_in[0];
  const float* w_qkv = (const float*)d_in[1];
  const float* b_qkv = (const float*)d_in[2];
  const float* w_out = (const float*)d_in[3];
  const float* b_out = (const float*)d_in[4];
  float* out = (float*)d_out;

  char* ws = (char*)d_ws;
  size_t off = 0;
  auto alloc = [&](size_t bytes) {
    size_t o = off;
    off += (bytes + 255) & ~(size_t)255;
    return o;
  };
  unsigned short* x16  = (unsigned short*)(ws + alloc((size_t)BT_ * E_ * 2));
  unsigned short* wq16 = (unsigned short*)(ws + alloc((size_t)E_ * N3E * 2));   // [3E][E]
  unsigned short* wo16 = (unsigned short*)(ws + alloc((size_t)E_ * E_ * 2));    // [E][E]
  float*          qkvf = (float*)(ws + alloc((size_t)BT_ * N3E * 4));
  unsigned short* q16  = (unsigned short*)(ws + alloc((size_t)B_ * H_ * T_ * D_ * 2));
  unsigned short* k16  = (unsigned short*)(ws + alloc((size_t)B_ * H_ * T_ * D_ * 2));
  unsigned short* vT16 = (unsigned short*)(ws + alloc((size_t)B_ * H_ * T_ * D_ * 2));
  float* attnf = qkvf;            // alias: qkv f32 dead after rope_pack
  unsigned short* attn16 = x16;   // alias: x16 dead after GEMM1

  k_f32_to_bf16<<<(BT_ * E_ / 4 + 255) / 256, 256, 0, stream>>>(x, x16, BT_ * E_);
  // weights: transpose+convert to [N][K] bf16 so GEMM staging is contiguous
  k_transpose_bf16<<<dim3(N3E / 32, E_ / 32), 256, 0, stream>>>(w_qkv, wq16, E_, N3E);
  k_transpose_bf16<<<dim3(E_ / 32, E_ / 32), 256, 0, stream>>>(w_out, wo16, E_, E_);

  k_gemm_bf16<<<dim3(N3E / 128, BT_ / 128), 256, 0, stream>>>(
      x16, wq16, b_qkv, qkvf, BT_, N3E, E_);

  k_rope_pack<<<(B_ * T_ * E_) / 256, 256, 0, stream>>>(qkvf, q16, k16, vT16);

  k_flash_attn<<<dim3(T_ / 64, H_, B_), 128, 0, stream>>>(q16, k16, vT16, attnf);

  k_f32_to_bf16<<<(BT_ * E_ / 4 + 255) / 256, 256, 0, stream>>>(attnf, attn16, BT_ * E_);

  k_gemm_bf16<<<dim3(E_ / 128, BT_ / 128), 256, 0, stream>>>(
      attn16, wo16, b_out, out, BT_, E_, E_);
}
